// SmileGIN_84799834292469
// MI455X (gfx1250) — matrix operand order repaired and checked by reading, not compiled, not run on hardware
//
#include <hip/hip_runtime.h>
#include <math.h>

#define N_NODES  524288
#define N_EDGES  2097152
#define N_GRAPHS 16384
#define HIDDEN   128

typedef __attribute__((ext_vector_type(16))) __bf16 v16bf;
typedef __attribute__((ext_vector_type(8)))  float  v8f;

__device__ inline __bf16 f2bf(float f) {
    union { float f; unsigned u; } v; v.f = f;
    unsigned r = v.u + 0x7FFFu + ((v.u >> 16) & 1u);   // round-to-nearest-even
    unsigned short h = (unsigned short)(r >> 16);
    __bf16 o; __builtin_memcpy(&o, &h, 2);
    return o;
}

__device__ inline void atomicMaxF(float* addr, float val) {
    unsigned* ua = (unsigned*)addr;
    unsigned old = *ua;
    while (__uint_as_float(old) < val) {
        unsigned assumed = old;
        old = atomicCAS(ua, assumed, __float_as_uint(val));
        if (old == assumed) break;
    }
}

template<int MODE, int KSRC>
__device__ inline void load_run8(const float* __restrict__ p0,
                                 const float* __restrict__ p1,
                                 int k, float* r) {
    const bool inb = (KSRC % 32 == 0) || (k < KSRC);   // folds away when K%32==0
    if (inb) {
        float4 x0 = *(const float4*)(p0 + k);
        float4 x1 = *(const float4*)(p0 + k + 4);
        if constexpr (MODE == 1) {  // A = agg + x
            float4 y0 = *(const float4*)(p1 + k);
            float4 y1 = *(const float4*)(p1 + k + 4);
            x0.x += y0.x; x0.y += y0.y; x0.z += y0.z; x0.w += y0.w;
            x1.x += y1.x; x1.y += y1.y; x1.z += y1.z; x1.w += y1.w;
        }
        r[0]=x0.x; r[1]=x0.y; r[2]=x0.z; r[3]=x0.w;
        r[4]=x1.x; r[5]=x1.y; r[6]=x1.z; r[7]=x1.w;
    } else {
        #pragma unroll
        for (int i = 0; i < 8; ++i) r[i] = 0.f;
    }
}

// One wave computes a 16x128 output tile with v_wmma_f32_16x16x32_bf16.
// MODE 0: out = relu(A0 @ W + b)                         (atom_proj)
// MODE 1: out = resid + relu((A0 + A1) @ W + b)          (GIN layer, in-place x)
// MODE 2: pool_out[batch[row]] += alpha[row]*relu(A0@W+b) (attentional pooling)
// MODE 3: out = A0 @ W + b                               (global branch, K=200 padded)
template<int KSRC, int MODE>
__global__ __launch_bounds__(256)
void gemm_wmma_kernel(const float* __restrict__ A0, const float* __restrict__ A1,
                      int astride,
                      const float* __restrict__ W, const float* __restrict__ bias,
                      float* __restrict__ out, const float* __restrict__ resid,
                      const float* __restrict__ alpha, const int* __restrict__ batch,
                      float* __restrict__ pool_out) {
    constexpr int KT   = (KSRC + 31) / 32;
    constexpr int KPAD = KT * 32;
    // W transposed to bf16 in LDS: Wt[col][k] -> each lane's B fragment is
    // 16 contiguous bf16 (32B) = 2x ds_load_b128.
    __shared__ __attribute__((aligned(32))) __bf16 Wt[HIDDEN * KPAD];

    const int tid = threadIdx.x;
    for (int idx = tid; idx < HIDDEN * KPAD; idx += 256) {
        const int col = idx / KPAD;
        const int k   = idx - col * KPAD;
        const float w = (k < KSRC) ? W[(size_t)k * HIDDEN + col] : 0.f;
        Wt[idx] = f2bf(w);                     // idx == col*KPAD + k
    }
    __syncthreads();

    const int wave = tid >> 5;
    const int lane = tid & 31;
    const int half = lane >> 4;                // 0: lanes 0-15, 1: lanes 16-31
    const int lm   = lane & 15;
    const int rowbase = blockIdx.x * 128 + wave * 16;

    const float* rp0 = A0 + (size_t)(rowbase + lm) * astride;
    const float* rp1 = (MODE == 1) ? (A1 + (size_t)(rowbase + lm) * astride) : nullptr;

    v8f acc[8] = {};

    #pragma unroll
    for (int kt = 0; kt < KT; ++kt) {
        // A 16x32 k order used here: lanes 0-15 hold K 0-15, lanes 16-31 hold K 16-31 (contiguous, matches B)
        const int k0 = kt * 32 + half * 16;
        const int k1 = k0 + 8;
        float r0[8], r1[8];
        load_run8<MODE, KSRC>(rp0, rp1, k0, r0);
        load_run8<MODE, KSRC>(rp0, rp1, k1, r1);
        v16bf a;
        #pragma unroll
        for (int i = 0; i < 8; ++i) { a[i] = f2bf(r0[i]); a[i + 8] = f2bf(r1[i]); }

        #pragma unroll
        for (int nt = 0; nt < 8; ++nt) {
            // B 32x16: lane (col = nt*16+lm) holds K kbase..kbase+15 contiguous in Wt
            const v16bf b = *(const v16bf*)(&Wt[(nt * 16 + lm) * KPAD + kt * 32 + half * 16]);
            acc[nt] = __builtin_amdgcn_wmma_f32_16x16x32_bf16(
                false, a, false, b, (short)0, acc[nt], false, false);
        }
    }

    float av8[8]; int g8[8];
    if constexpr (MODE == 2) {
        #pragma unroll
        for (int r = 0; r < 8; ++r) {
            const int row = rowbase + r + half * 8;
            av8[r] = alpha[row];
            g8[r]  = batch[row];
        }
    }

    #pragma unroll
    for (int nt = 0; nt < 8; ++nt) {
        const int col = nt * 16 + lm;
        const float bb = bias[col];
        #pragma unroll
        for (int r = 0; r < 8; ++r) {
            const int row = rowbase + r + half * 8;   // D: VGPR r -> rows r / r+8
            const float v = acc[nt][r] + bb;
            if constexpr (MODE == 0) {
                out[(size_t)row * HIDDEN + col] = fmaxf(v, 0.f);
            } else if constexpr (MODE == 1) {
                out[(size_t)row * HIDDEN + col] =
                    resid[(size_t)row * HIDDEN + col] + fmaxf(v, 0.f);
            } else if constexpr (MODE == 2) {
                atomicAdd(&pool_out[(size_t)g8[r] * HIDDEN + col], av8[r] * fmaxf(v, 0.f));
            } else {
                out[(size_t)row * HIDDEN + col] = v;
            }
        }
    }
}

// one wave per edge: float4 gather from x[src], 4 global f32 atomic adds into agg[dst]
__global__ __launch_bounds__(256)
void scatter_add_kernel(const int* __restrict__ ei, const float* __restrict__ x,
                        float* __restrict__ agg) {
    const int wave = threadIdx.x >> 5, lane = threadIdx.x & 31;
    const int e = blockIdx.x * 8 + wave;
    const int src = ei[e];
    const int dst = ei[N_EDGES + e];
    const float4 v = *(const float4*)(x + (size_t)src * HIDDEN + lane * 4);
    float* d = agg + (size_t)dst * HIDDEN + lane * 4;
    atomicAdd(d + 0, v.x); atomicAdd(d + 1, v.y);
    atomicAdd(d + 2, v.z); atomicAdd(d + 3, v.w);
}

template<bool RELU>
__global__ __launch_bounds__(256)
void layernorm_kernel(float* __restrict__ io, const float* __restrict__ g,
                      const float* __restrict__ b) {
    const int wave = threadIdx.x >> 5, lane = threadIdx.x & 31;
    const size_t row = (size_t)blockIdx.x * 8 + wave;
    float v[4], s = 0.f, ss = 0.f;
    #pragma unroll
    for (int i = 0; i < 4; ++i) {
        v[i] = io[row * HIDDEN + lane + 32 * i];
        s += v[i]; ss += v[i] * v[i];
    }
    #pragma unroll
    for (int m = 16; m >= 1; m >>= 1) {
        s  += __shfl_xor(s, m, 32);
        ss += __shfl_xor(ss, m, 32);
    }
    const float mu  = s * (1.f / 128.f);
    const float var = ss * (1.f / 128.f) - mu * mu;
    const float rs  = rsqrtf(var + 1e-5f);
    #pragma unroll
    for (int i = 0; i < 4; ++i) {
        const int col = lane + 32 * i;
        float o = (v[i] - mu) * rs * g[col] + b[col];
        if (RELU) o = fmaxf(o, 0.f);
        io[row * HIDDEN + col] = o;
    }
}

__global__ __launch_bounds__(256)
void gate_kernel(const float* __restrict__ x, const float* __restrict__ Wg,
                 const float* __restrict__ bg, const int* __restrict__ batch,
                 float* __restrict__ gate, float* __restrict__ m) {
    const int wave = threadIdx.x >> 5, lane = threadIdx.x & 31;
    const size_t n = (size_t)blockIdx.x * 8 + wave;
    float s = 0.f;
    #pragma unroll
    for (int i = 0; i < 4; ++i) {
        const int c = lane + 32 * i;
        s += x[n * HIDDEN + c] * Wg[c];
    }
    #pragma unroll
    for (int msk = 16; msk >= 1; msk >>= 1) s += __shfl_xor(s, msk, 32);
    if (lane == 0) {
        const float gv = s + bg[0];
        gate[n] = gv;
        atomicMaxF(&m[batch[n]], gv);
    }
}

__global__ __launch_bounds__(256)
void softmax_e_kernel(const float* __restrict__ gate, const float* __restrict__ m,
                      const int* __restrict__ batch, float* __restrict__ e,
                      float* __restrict__ denom) {
    const size_t n = (size_t)blockIdx.x * 256 + threadIdx.x;
    const int g = batch[n];
    const float ev = expf(gate[n] - m[g]);
    e[n] = ev;
    atomicAdd(&denom[g], ev);
}

__global__ __launch_bounds__(256)
void alpha_kernel(const float* __restrict__ e, const float* __restrict__ denom,
                  const int* __restrict__ batch, float* __restrict__ alpha) {
    const size_t n = (size_t)blockIdx.x * 256 + threadIdx.x;
    alpha[n] = e[n] / denom[batch[n]];
}

__global__ __launch_bounds__(64)
void head_kernel(const float* __restrict__ gf, const float* __restrict__ glf,
                 const float* __restrict__ We, const float* __restrict__ be,
                 const float* __restrict__ W1, const float* __restrict__ b1,
                 const float* __restrict__ W2, const float* __restrict__ b2,
                 float* __restrict__ out) {
    __shared__ float sm[3 * HIDDEN];
    __shared__ float semb[64];
    __shared__ float st[64];
    const int g = blockIdx.x, tid = threadIdx.x;
    for (int i = tid; i < HIDDEN; i += 64) {
        const float a = gf[(size_t)g * HIDDEN + i];
        const float c = glf[(size_t)g * HIDDEN + i];
        sm[i] = a; sm[HIDDEN + i] = c; sm[2 * HIDDEN + i] = a * c;
    }
    __syncthreads();
    float acc = be[tid];
    for (int i = 0; i < 3 * HIDDEN; ++i) acc += sm[i] * We[i * 64 + tid];
    semb[tid] = acc;                         // emb: no activation
    __syncthreads();
    float h = b1[tid];
    for (int i = 0; i < 64; ++i) h += semb[i] * W1[i * 64 + tid];
    h = fmaxf(h, 0.f);
    st[tid] = h * W2[tid];
    __syncthreads();
    for (int s = 32; s >= 1; s >>= 1) {
        if (tid < s) st[tid] += st[tid + s];
        __syncthreads();
    }
    if (tid == 0) out[g] = st[0] + b2[0];
}

__global__ __launch_bounds__(256)
void zero4_kernel(float4* __restrict__ p) {
    p[(size_t)blockIdx.x * 256 + threadIdx.x] = make_float4(0.f, 0.f, 0.f, 0.f);
}

__global__ __launch_bounds__(256)
void fill_kernel(float* __restrict__ p, float v) {
    p[(size_t)blockIdx.x * 256 + threadIdx.x] = v;
}

extern "C" void kernel_launch(void* const* d_in, const int* in_sizes, int n_in,
                              void* d_out, int out_size, void* d_ws, size_t ws_size,
                              hipStream_t stream) {
    const float* graph_x    = (const float*)d_in[0];
    const float* global_x   = (const float*)d_in[1];
    const int*   edge_index = (const int*)d_in[2];
    const int*   batch      = (const int*)d_in[3];

    const float *atomW,*atomB,*gin0W,*gin0B,*gin1W,*gin1B,*normG,*normB,*gateW,*gateB,
                *poolW,*poolB,*globW,*globB,*gnG,*gnB,*embW,*embB,*h1W,*h1B,*h2W,*h2B;
    #define P(i) ((const float*)d_in[i])
    if (n_in > 6 && in_sizes[6] == 24576) {
        // jax pytree flatten: dict keys sorted alphabetically
        atomW=P(4);  atomB=P(5);  embW=P(6);   embB=P(7);  gateW=P(8);  gateB=P(9);
        gin0W=P(10); gin0B=P(11); gin1W=P(12); gin1B=P(13); globW=P(14); globB=P(15);
        gnG=P(16);   gnB=P(17);   h1W=P(18);   h1B=P(19);  h2W=P(20);   h2B=P(21);
        normG=P(22); normB=P(23); poolW=P(24); poolB=P(25);
    } else {
        // dict insertion order
        atomW=P(4);  atomB=P(5);  gin0W=P(6);  gin0B=P(7); gin1W=P(8);  gin1B=P(9);
        normG=P(10); normB=P(11); gateW=P(12); gateB=P(13); poolW=P(14); poolB=P(15);
        globW=P(16); globB=P(17); gnG=P(18);   gnB=P(19);  embW=P(20);  embB=P(21);
        h1W=P(22);   h1B=P(23);   h2W=P(24);   h2B=P(25);
    }
    #undef P

    // workspace: x (256 MiB) | agg (256 MiB, reused for pooling scratch after GIN)
    char* ws = (char*)d_ws;
    float* x   = (float*)ws;
    float* agg = (float*)(ws + (size_t)N_NODES * HIDDEN * sizeof(float));
    float* gate   = agg;
    float* ebuf   = agg + (size_t)N_NODES;
    float* alph   = agg + 2 * (size_t)N_NODES;
    float* mbuf   = agg + 3 * (size_t)N_NODES;
    float* denom  = mbuf + N_GRAPHS;
    float* gfeat  = denom + N_GRAPHS;
    float* glfeat = gfeat + (size_t)N_GRAPHS * HIDDEN;
    float* outp   = (float*)d_out;

    // 1) x = relu(graph_x @ Wa + ba)          [N,64]x[64,128]
    gemm_wmma_kernel<64, 0><<<N_NODES / 128, 256, 0, stream>>>(
        graph_x, nullptr, 64, atomW, atomB, x, nullptr, nullptr, nullptr, nullptr);

    // 2) two GIN layers: x = x + relu((segment_sum(x[src],dst) + x) @ W + b)
    const float* gW[2] = {gin0W, gin1W};
    const float* gB[2] = {gin0B, gin1B};
    for (int l = 0; l < 2; ++l) {
        zero4_kernel<<<(N_NODES * HIDDEN / 4) / 256, 256, 0, stream>>>((float4*)agg);
        scatter_add_kernel<<<N_EDGES / 8, 256, 0, stream>>>(edge_index, x, agg);
        gemm_wmma_kernel<128, 1><<<N_NODES / 128, 256, 0, stream>>>(
            agg, x, 128, gW[l], gB[l], x, x, nullptr, nullptr, nullptr);
    }

    // 3) node layernorm (in place)
    layernorm_kernel<false><<<N_NODES / 8, 256, 0, stream>>>(x, normG, normB);

    // 4) attentional aggregation
    fill_kernel<<<N_GRAPHS / 256, 256, 0, stream>>>(mbuf, -INFINITY);
    fill_kernel<<<N_GRAPHS / 256, 256, 0, stream>>>(denom, 0.f);
    zero4_kernel<<<(N_GRAPHS * HIDDEN / 4) / 256, 256, 0, stream>>>((float4*)gfeat);
    gate_kernel<<<N_NODES / 8, 256, 0, stream>>>(x, gateW, gateB, batch, gate, mbuf);
    softmax_e_kernel<<<N_NODES / 256, 256, 0, stream>>>(gate, mbuf, batch, ebuf, denom);
    alpha_kernel<<<N_NODES / 256, 256, 0, stream>>>(ebuf, denom, batch, alph);
    gemm_wmma_kernel<128, 2><<<N_NODES / 128, 256, 0, stream>>>(
        x, nullptr, 128, poolW, poolB, nullptr, nullptr, alph, batch, gfeat);

    // 5) global branch: relu(layernorm(global_x @ Wgl + bgl))  (K=200 zero-padded to 224)
    gemm_wmma_kernel<200, 3><<<N_GRAPHS / 128, 256, 0, stream>>>(
        global_x, nullptr, 200, globW, globB, glfeat, nullptr, nullptr, nullptr, nullptr);
    layernorm_kernel<true><<<N_GRAPHS / 8, 256, 0, stream>>>(glfeat, gnG, gnB);

    // 6) fusion + MLP head -> out[G]
    head_kernel<<<N_GRAPHS, 64, 0, stream>>>(
        gfeat, glfeat, embW, embB, h1W, h1B, h2W, h2B, outp);
}
